// MotorLearningRNN_18141941858971
// MI455X (gfx1250) — matrix operand order repaired and checked by reading, not compiled, not run on hardware
//
#include <hip/hip_runtime.h>
#include <cstdint>
#include <cstddef>

// ---------------------------------------------------------------------------
// MotorLearningRNN on MI455X (gfx1250, wave32, WMMA + async global->LDS DMA)
//
// 49 sequential RNN steps of  h' = tanh(h @ Whh^T + xin @ Wih^T) with
// B=4096, H=2048. Main GEMM runs on v_wmma_f32_16x16x32_bf16 using a
// split-precision (bf16 hi + bf16 lo) representation of both h and Whh
// (3 WMMA products per tile for ~fp32 accuracy). Tiles are streamed into
// LDS with global_load_async_to_lds_b128 (ASYNCcnt), double-buffered.
// ---------------------------------------------------------------------------

#define B_DIM   4096
#define H_DIM   2048
#define T_DIM   7
#define I_DIM   7
#define O_DIM   4
#define R_RUNS  7

typedef __bf16 bf16;
typedef __attribute__((ext_vector_type(16))) __bf16 v16bf;
typedef __attribute__((ext_vector_type(8)))  __bf16 v8bf;
typedef __attribute__((ext_vector_type(8)))  float  v8f;

// workspace layout (bytes)
static constexpr size_t OFF_WHI  = 0;                         // Whh hi  [H*H] bf16  (8 MiB)
static constexpr size_t OFF_WLO  = 8u  * 1024 * 1024;         // Whh lo  [H*H] bf16  (8 MiB)
static constexpr size_t OFF_HHI0 = 16u * 1024 * 1024;         // h hi buf0 [B*H] bf16 (16 MiB)
static constexpr size_t OFF_HLO0 = 32u * 1024 * 1024;         // h lo buf0
static constexpr size_t OFF_HHI1 = 48u * 1024 * 1024;         // h hi buf1
static constexpr size_t OFF_HLO1 = 64u * 1024 * 1024;         // h lo buf1
static constexpr size_t OFF_XIN0 = 80u * 1024 * 1024;         // xin buf0 [B*T*I] f32
static constexpr size_t OFF_XIN1 = OFF_XIN0 + (size_t)B_DIM * T_DIM * I_DIM * 4;

__device__ __forceinline__ v16bf frag_join(v8bf a, v8bf b) {
  return __builtin_shufflevector(a, b, 0,1,2,3,4,5,6,7,8,9,10,11,12,13,14,15);
}
__device__ __forceinline__ v8bf lds_ld8(const bf16* p) { return *(const v8bf*)p; }

// Async DMA: 16 bytes global -> LDS, per lane. Tracked by ASYNCcnt.
__device__ __forceinline__ void async_b128(uint32_t lds_off, const void* gptr) {
  asm volatile("global_load_async_to_lds_b128 %0, %1, off"
               :: "v"(lds_off), "v"((unsigned long long)(uintptr_t)gptr)
               : "memory");
}
__device__ __forceinline__ void wait_async0() {
  asm volatile("s_wait_asynccnt 0x0" ::: "memory");
}
// low 32 bits of a flat LDS address == workgroup-relative LDS byte offset
__device__ __forceinline__ uint32_t lds_addr32(const void* p) {
  return (uint32_t)(uintptr_t)p;
}

// ---------------------------------------------------------------------------
// Prep: split fp32 weights into bf16 hi + lo (kept in original [j][k] layout:
// a WMMA B-fragment for column j of Whh^T is a contiguous row segment of Whh).
// ---------------------------------------------------------------------------
__global__ void split_kernel(const float* __restrict__ w,
                             bf16* __restrict__ hi, bf16* __restrict__ lo, int n) {
  int i = blockIdx.x * 256 + threadIdx.x;
  if (i < n) {
    float v = w[i];
    bf16  h = (bf16)v;
    hi[i] = h;
    lo[i] = (bf16)(v - (float)h);
  }
}

__global__ void zero16_kernel(uint4* __restrict__ p, long n16) {
  long i = (long)blockIdx.x * 256 + threadIdx.x;
  if (i < n16) p[i] = make_uint4(0u, 0u, 0u, 0u);
}

// ---------------------------------------------------------------------------
// One RNN step:  Hout = tanh(Hin @ Whh^T + xin[:,t,:] @ Wih^T + b_ih + b_hh)
// C-tile 128x128 per block (256 thr = 8 waves; wave = 4x2 WMMA tiles = 64x32).
// ---------------------------------------------------------------------------
#define LDK 40   // 32 k-elements + 8 pad (80 B rows -> conflict-free frags)

__global__ __launch_bounds__(256)
void rnn_step_kernel(const bf16* __restrict__ Hhi_in, const bf16* __restrict__ Hlo_in,
                     bf16* __restrict__ Hhi_out,      bf16* __restrict__ Hlo_out,
                     const bf16* __restrict__ Whi,    const bf16* __restrict__ Wlo,
                     const float* __restrict__ xin, int tstep,
                     const float* __restrict__ W_ih,
                     const float* __restrict__ b_ih,  const float* __restrict__ b_hh) {
  __shared__ __align__(16) bf16 sA[2][2][128 * LDK];  // [buf][hi/lo][row*LDK+k]
  __shared__ __align__(16) bf16 sB[2][2][128 * LDK];
  __shared__ float sX[128][8];   // xin rows for this 128-row tile
  __shared__ float sW[128][8];   // W_ih rows (+ fused bias in [7]) for this 128-col tile

  const int tid  = threadIdx.x;
  const int lane = tid & 31;
  const int wave = tid >> 5;
  const int wM   = wave & 1;          // 2 waves along M (64 rows each)
  const int wN   = wave >> 1;         // 4 waves along N (32 cols each)
  const int lrow = lane & 15;
  const int half = lane >> 4;

  const int rowBase = blockIdx.y * 128;   // batch rows
  const int colBase = blockIdx.x * 128;   // hidden cols (j)

  // per-thread DMA slice: two 16B chunks per array per k-chunk
  const int c0   = tid;            // chunk ids
  const int c1   = tid + 256;
  const int row0 = c0 >> 2, kp0 = c0 & 3;
  const int row1 = c1 >> 2, kp1 = c1 & 3;

  auto async_tile = [&](int buf, int k0) {
    {
      const size_t ao = ((size_t)(rowBase + row0) << 11) + (size_t)(k0 + kp0 * 8);
      const size_t bo = ((size_t)(colBase + row0) << 11) + (size_t)(k0 + kp0 * 8);
      const int off = row0 * LDK + kp0 * 8;
      async_b128(lds_addr32(&sA[buf][0][off]), Hhi_in + ao);
      async_b128(lds_addr32(&sA[buf][1][off]), Hlo_in + ao);
      async_b128(lds_addr32(&sB[buf][0][off]), Whi   + bo);
      async_b128(lds_addr32(&sB[buf][1][off]), Wlo   + bo);
    }
    {
      const size_t ao = ((size_t)(rowBase + row1) << 11) + (size_t)(k0 + kp1 * 8);
      const size_t bo = ((size_t)(colBase + row1) << 11) + (size_t)(k0 + kp1 * 8);
      const int off = row1 * LDK + kp1 * 8;
      async_b128(lds_addr32(&sA[buf][0][off]), Hhi_in + ao);
      async_b128(lds_addr32(&sA[buf][1][off]), Hlo_in + ao);
      async_b128(lds_addr32(&sB[buf][0][off]), Whi   + bo);
      async_b128(lds_addr32(&sB[buf][1][off]), Wlo   + bo);
    }
  };

  // kick off first tile DMA, then stage the tiny xin / W_ih slices + biases
  async_tile(0, 0);
  if (tid < 128) {
    const int b = rowBase + tid;
    #pragma unroll
    for (int i = 0; i < I_DIM; i++)
      sX[tid][i] = xin[((size_t)b * T_DIM + tstep) * I_DIM + i];
    sX[tid][7] = 0.0f;
  } else {
    const int r = tid - 128;
    const int j = colBase + r;
    #pragma unroll
    for (int i = 0; i < I_DIM; i++)
      sW[r][i] = W_ih[j * I_DIM + i];
    sW[r][7] = b_ih[j] + b_hh[j];
  }
  wait_async0();
  __syncthreads();

  v8f acc[4][2] = {};

  const int KT = H_DIM / 32;  // 64 k-chunks
  for (int kt = 0; kt < KT; kt++) {
    const int nxt = kt + 1;
    if (nxt < KT) async_tile(nxt & 1, nxt * 32);   // DMA next chunk while computing

    const int buf = kt & 1;
    const bf16* Ah = &sA[buf][0][0];
    const bf16* Al = &sA[buf][1][0];
    const bf16* Bh = &sB[buf][0][0];
    const bf16* Bl = &sB[buf][1][0];

    // B fragments: column j = cb + lrow; K chunks at half*8 and half*8+16
    v16bf bHi[2], bLo[2];
    #pragma unroll
    for (int ni = 0; ni < 2; ni++) {
      const int j   = wN * 32 + ni * 16 + lrow;
      const int off = j * LDK + half * 8;
      bHi[ni] = frag_join(lds_ld8(Bh + off), lds_ld8(Bh + off + 16));
      bLo[ni] = frag_join(lds_ld8(Bl + off), lds_ld8(Bl + off + 16));
    }

    #pragma unroll
    for (int mi = 0; mi < 4; mi++) {
      // A fragments: row = ra + lrow; K chunks at half*8 and half*8+16
      const int ra  = wM * 64 + mi * 16 + lrow;
      const int off = ra * LDK + half * 8;
      v16bf aHi = frag_join(lds_ld8(Ah + off), lds_ld8(Ah + off + 16));
      v16bf aLo = frag_join(lds_ld8(Al + off), lds_ld8(Al + off + 16));
      #pragma unroll
      for (int ni = 0; ni < 2; ni++) {
        acc[mi][ni] = __builtin_amdgcn_wmma_f32_16x16x32_bf16(
            false, aHi, false, bHi[ni], (short)0, acc[mi][ni], false, false);
        acc[mi][ni] = __builtin_amdgcn_wmma_f32_16x16x32_bf16(
            false, aHi, false, bLo[ni], (short)0, acc[mi][ni], false, false);
        acc[mi][ni] = __builtin_amdgcn_wmma_f32_16x16x32_bf16(
            false, aLo, false, bHi[ni], (short)0, acc[mi][ni], false, false);
      }
    }

    if (nxt < KT) {
      wait_async0();     // this wave's DMA for chunk nxt has landed in LDS
      __syncthreads();   // all waves' DMA visible; also fences reads of buf before it is refilled
    }
  }

  // epilogue: + xin@Wih^T + biases, tanh, split-bf16 writeback
  #pragma unroll
  for (int mi = 0; mi < 4; mi++) {
    #pragma unroll
    for (int ni = 0; ni < 2; ni++) {
      #pragma unroll
      for (int r = 0; r < 8; r++) {
        const int lr = wM * 64 + mi * 16 + half * 8 + r;  // local row
        const int lc = wN * 32 + ni * 16 + lrow;          // local col
        float c = acc[mi][ni][r];
        #pragma unroll
        for (int i = 0; i < I_DIM; i++) c += sX[lr][i] * sW[lc][i];
        c += sW[lc][7];
        const float h = tanhf(c);
        const bf16 hi = (bf16)h;
        const bf16 lo = (bf16)(h - (float)hi);
        const size_t g = ((size_t)(rowBase + lr) << 11) + (size_t)(colBase + lc);
        Hhi_out[g] = hi;
        Hlo_out[g] = lo;
      }
    }
  }
}

// ---------------------------------------------------------------------------
// Head: y = h @ W_fc^T + b_fc  -> out[:, col, :]
//       xin_next[:, t, :] = y @ W_map^T + b_map   (skipped on last run)
// One wave per batch row; cross-lane reduce over H.
// ---------------------------------------------------------------------------
__global__ __launch_bounds__(256)
void rnn_head_kernel(const bf16* __restrict__ Hhi, const bf16* __restrict__ Hlo,
                     const float* __restrict__ W_fc, const float* __restrict__ b_fc,
                     const float* __restrict__ W_map, const float* __restrict__ b_map,
                     float* __restrict__ out, int col,
                     float* __restrict__ xin_next, int tstep, int writeXin) {
  const int lane = threadIdx.x & 31;
  const int wave = threadIdx.x >> 5;
  const int b    = blockIdx.x * 8 + wave;

  const bf16* ph = Hhi + ((size_t)b << 11);
  const bf16* pl = Hlo + ((size_t)b << 11);

  float s0 = 0.f, s1 = 0.f, s2 = 0.f, s3 = 0.f;
  for (int k = lane; k < H_DIM; k += 32) {
    const float h = (float)ph[k] + (float)pl[k];
    s0 += h * W_fc[k];
    s1 += h * W_fc[H_DIM + k];
    s2 += h * W_fc[2 * H_DIM + k];
    s3 += h * W_fc[3 * H_DIM + k];
  }
  #pragma unroll
  for (int off = 16; off > 0; off >>= 1) {
    s0 += __shfl_xor(s0, off, 32);
    s1 += __shfl_xor(s1, off, 32);
    s2 += __shfl_xor(s2, off, 32);
    s3 += __shfl_xor(s3, off, 32);
  }
  if (lane == 0) {
    const float y0 = s0 + b_fc[0];
    const float y1 = s1 + b_fc[1];
    const float y2 = s2 + b_fc[2];
    const float y3 = s3 + b_fc[3];
    float* po = out + ((size_t)b * (R_RUNS * T_DIM) + col) * O_DIM;
    po[0] = y0; po[1] = y1; po[2] = y2; po[3] = y3;
    if (writeXin) {
      #pragma unroll
      for (int i = 0; i < I_DIM; i++) {
        const float v = b_map[i] + y0 * W_map[i * O_DIM + 0] + y1 * W_map[i * O_DIM + 1]
                                 + y2 * W_map[i * O_DIM + 2] + y3 * W_map[i * O_DIM + 3];
        xin_next[((size_t)b * T_DIM + tstep) * I_DIM + i] = v;
      }
    }
  }
}

// ---------------------------------------------------------------------------
extern "C" void kernel_launch(void* const* d_in, const int* in_sizes, int n_in,
                              void* d_out, int out_size, void* d_ws, size_t ws_size,
                              hipStream_t stream) {
  (void)in_sizes; (void)n_in; (void)out_size; (void)ws_size;
  const float* x     = (const float*)d_in[0];
  const float* W_ih  = (const float*)d_in[1];
  const float* W_hh  = (const float*)d_in[2];
  const float* b_ih  = (const float*)d_in[3];
  const float* b_hh  = (const float*)d_in[4];
  const float* W_fc  = (const float*)d_in[5];
  const float* b_fc  = (const float*)d_in[6];
  const float* W_map = (const float*)d_in[7];
  const float* b_map = (const float*)d_in[8];
  float* out = (float*)d_out;

  char* ws = (char*)d_ws;
  bf16* Whi = (bf16*)(ws + OFF_WHI);
  bf16* Wlo = (bf16*)(ws + OFF_WLO);
  bf16* Hhi[2] = { (bf16*)(ws + OFF_HHI0), (bf16*)(ws + OFF_HHI1) };
  bf16* Hlo[2] = { (bf16*)(ws + OFF_HLO0), (bf16*)(ws + OFF_HLO1) };
  float* xinbuf[2] = { (float*)(ws + OFF_XIN0), (float*)(ws + OFF_XIN1) };

  // 1) split Whh into bf16 hi/lo
  {
    const int n = H_DIM * H_DIM;
    split_kernel<<<(n + 255) / 256, 256, 0, stream>>>(W_hh, Whi, Wlo, n);
  }
  // 2) zero h0 (hi+lo buf0 are contiguous: 32 MiB)
  {
    const long n16 = (2L * (long)B_DIM * H_DIM * sizeof(bf16)) / 16;
    zero16_kernel<<<(unsigned)((n16 + 255) / 256), 256, 0, stream>>>(
        (uint4*)(ws + OFF_HHI0), n16);
  }

  // 3) 49 sequential steps (runs fused; last run's trailing RNN is dead code)
  const dim3 stepGrid(H_DIM / 128, B_DIM / 128);  // 16 x 32
  for (int s = 0; s < R_RUNS * T_DIM; s++) {
    const int r = s / T_DIM;
    const int t = s % T_DIM;
    const int ri = s & 1;
    const int wi = 1 - ri;
    const float* xin = (r == 0) ? x : xinbuf[(r - 1) & 1];

    rnn_step_kernel<<<stepGrid, 256, 0, stream>>>(
        Hhi[ri], Hlo[ri], Hhi[wi], Hlo[wi], Whi, Wlo,
        xin, t, W_ih, b_ih, b_hh);

    rnn_head_kernel<<<B_DIM / 8, 256, 0, stream>>>(
        Hhi[wi], Hlo[wi], W_fc, b_fc, W_map, b_map,
        out, r * T_DIM + t, xinbuf[r & 1], t, (r < R_RUNS - 1) ? 1 : 0);
  }
}
